// GRACEModel_32229434589361
// MI455X (gfx1250) — compile-verified, matrix operand order checked by reading, not run on hardware
//
#include <hip/hip_runtime.h>

// ---------------- types ----------------
typedef __bf16 v16bf __attribute__((ext_vector_type(16)));
typedef __bf16 v8bf  __attribute__((ext_vector_type(8)));
typedef float  v8f   __attribute__((ext_vector_type(8)));

#define NROWS 8192
#define DIM   256   // D == P == 256

__device__ __forceinline__ v8f wmma_bf16(v16bf a, v16bf b, v8f c) {
  return __builtin_amdgcn_wmma_f32_16x16x32_bf16(
      /*neg_a=*/false, a, /*neg_b=*/false, b,
      /*c_mod=*/(short)0, c, /*reuse_a=*/false, /*reuse_b=*/false);
}

// Load one 16x32 bf16 operand fragment for the wave.
// Row-major source with row stride DIM. Per ISA layout:
// lane L holds row (base + L%16); lanes 0-15 hold K {kc*32+0..7, kc*32+16..23},
// lanes 16-31 hold K {kc*32+8..15, kc*32+24..31}.
__device__ __forceinline__ v16bf load_frag(const __bf16* base, int row, int kc, int h) {
  const __bf16* p = base + row * DIM + kc * 32 + 8 * h;
  v8bf lo = *(const v8bf*)p;
  v8bf hi = *(const v8bf*)(p + 16);
  return __builtin_shufflevector(lo, hi, 0,1,2,3,4,5,6,7,8,9,10,11,12,13,14,15);
}

// ---------------- elementwise convert ----------------
__global__ void cvt_f32_bf16(const float* __restrict__ s, __bf16* __restrict__ d, int n) {
  int i = blockIdx.x * blockDim.x + threadIdx.x;
  if (i < n) d[i] = (__bf16)s[i];
}

__global__ void zero_f32(float* __restrict__ p, int n) {
  int i = blockIdx.x * blockDim.x + threadIdx.x;
  if (i < n) p[i] = 0.0f;
}

// ---------------- projection GEMM: Out = act(A @ W^T + bias) ----------------
// A: 8192x256 bf16 row-major, W: 256x256 bf16 row-major (out col j <- W row j).
// Block = 128 threads (4 waves), tile 32x32, K = 256 in 8 chunks of 32.
template<bool ELU, bool BF16OUT>
__global__ __launch_bounds__(128) void proj_gemm(
    const __bf16* __restrict__ A0, const __bf16* __restrict__ A1,
    const __bf16* __restrict__ W,  const float* __restrict__ bias,
    void* __restrict__ O0, void* __restrict__ O1) {
  const __bf16* A = blockIdx.z ? A1 : A0;
  void* O = blockIdx.z ? O1 : O0;

  int tid  = threadIdx.x;
  int lane = tid & 31, wave = tid >> 5;
  int m = wave & 1, n = wave >> 1;
  int l15 = lane & 15, h = lane >> 4;

  int rA = blockIdx.x * 32 + 16 * m + l15;  // output row this lane feeds as A-operand
  int rW = blockIdx.y * 32 + 16 * n + l15;  // W row == output column

  v8f c = {0.f,0.f,0.f,0.f,0.f,0.f,0.f,0.f};
#pragma unroll
  for (int kc = 0; kc < 8; ++kc) {
    v16bf fa = load_frag(A, rA, kc, h);
    v16bf fw = load_frag(W, rW, kc, h);
    c = wmma_bf16(fa, fw, c);
  }

  int Cg = blockIdx.y * 32 + 16 * n + l15;   // C layout: col = lane%16
  float bv = bias[Cg];
#pragma unroll
  for (int v = 0; v < 8; ++v) {
    int Rg = blockIdx.x * 32 + 16 * m + v + 8 * h;  // C layout: row = v + 8*(lane/16)
    float x = c[v] + bv;
    if (ELU) x = (x > 0.0f) ? x : (__expf(x) - 1.0f);
    if (BF16OUT) ((__bf16*)O)[Rg * DIM + Cg] = (__bf16)x;
    else         ((float*)O)[Rg * DIM + Cg]  = x;
  }
}

// ---------------- row normalize + bf16 ----------------
__global__ __launch_bounds__(256) void normalize_rows(
    const float* __restrict__ G0, const float* __restrict__ G1,
    __bf16* __restrict__ O0, __bf16* __restrict__ O1) {
  const float* G = blockIdx.y ? G1 : G0;
  __bf16* O = blockIdx.y ? O1 : O0;
  int row = blockIdx.x, t = threadIdx.x;
  float v = G[row * DIM + t];
  __shared__ float red[256];
  red[t] = v * v;
  __syncthreads();
  for (int s = 128; s > 0; s >>= 1) {
    if (t < s) red[t] += red[t + s];
    __syncthreads();
  }
  float sc = 1.0f / fmaxf(sqrtf(red[0]), 1e-12f);
  O[row * DIM + t] = (__bf16)(v * sc);
}

// ---------------- fused similarity + exp + reductions ----------------
// Each block owns 32 rows (panel in LDS), loops over all 8192 cols in 64-steps.
// 4 waves = 4 column subtiles (n = wave). Each wave computes BOTH 16-row
// subtiles (m = 0,1) so every aJ/bJ fragment pair feeds 6 WMMAs (2x the
// arithmetic intensity vs L2 of a 1-m layout, and no duplicate global loads).
// Computes exp(2*aa), exp(2*ab), exp(2*bb); row-sums in registers,
// ab col-sums in LDS, diag(exp(2*ab)) to global.
__global__ __launch_bounds__(128) void sim_kernel(
    const __bf16* __restrict__ A, const __bf16* __restrict__ B,
    float* __restrict__ rsaa, float* __restrict__ rsab,
    float* __restrict__ csab, float* __restrict__ rsbb,
    float* __restrict__ dab) {
  __shared__ __bf16 sA[32 * DIM];
  __shared__ __bf16 sB[32 * DIM];
  __shared__ float  sCol[NROWS];

  int tid = threadIdx.x;
  int r0 = blockIdx.x * 32;

  {  // stage row panels (reused across all 128 column iterations)
    const float4* gA = (const float4*)(const void*)(A + r0 * DIM);
    const float4* gB = (const float4*)(const void*)(B + r0 * DIM);
    float4* lA = (float4*)(void*)sA;
    float4* lB = (float4*)(void*)sB;
    for (int i = tid; i < (32 * DIM) / 8; i += 128) { lA[i] = gA[i]; lB[i] = gB[i]; }
  }
  for (int i = tid; i < NROWS; i += 128) sCol[i] = 0.0f;
  __syncthreads();

  int lane = tid & 31;
  int n = tid >> 5;                 // column subtile owned by this wave
  int l15 = lane & 15, h = lane >> 4;

  const v8f Z = {0.f,0.f,0.f,0.f,0.f,0.f,0.f,0.f};
  v8f rsA0 = Z, rsA1 = Z, rsAB0 = Z, rsAB1 = Z, rsB0 = Z, rsB1 = Z;

  for (int c0 = 0; c0 < NROWS; c0 += 64) {
    int colRow = c0 + 16 * n + l15;  // global row of A/B used as B-operand (output col)
    v8f caa0 = Z, caa1 = Z, cab0 = Z, cab1 = Z, cbb0 = Z, cbb1 = Z;
#pragma unroll
    for (int kc = 0; kc < 8; ++kc) {
      v16bf aJ  = load_frag(A, colRow, kc, h);
      v16bf bJ  = load_frag(B, colRow, kc, h);
      v16bf aI0 = load_frag(sA, l15, kc, h);
      v16bf aI1 = load_frag(sA, 16 + l15, kc, h);
      v16bf bI0 = load_frag(sB, l15, kc, h);
      v16bf bI1 = load_frag(sB, 16 + l15, kc, h);
      caa0 = wmma_bf16(aI0, aJ, caa0);   // a @ a^T, rows 0..15
      caa1 = wmma_bf16(aI1, aJ, caa1);   // a @ a^T, rows 16..31
      cab0 = wmma_bf16(aI0, bJ, cab0);   // a @ b^T
      cab1 = wmma_bf16(aI1, bJ, cab1);
      cbb0 = wmma_bf16(bI0, bJ, cbb0);   // b @ b^T
      cbb1 = wmma_bf16(bI1, bJ, cbb1);
    }
    float colpart = 0.0f;
    int Cg = c0 + 16 * n + l15;
#pragma unroll
    for (int v = 0; v < 8; ++v) {
      int Rg0 = r0 + v + 8 * h;        // m = 0 rows
      int Rg1 = Rg0 + 16;              // m = 1 rows
      float eaa0 = __expf(2.0f * caa0[v]);  // exp(s / tau), tau = 0.5
      float eaa1 = __expf(2.0f * caa1[v]);
      float eab0 = __expf(2.0f * cab0[v]);
      float eab1 = __expf(2.0f * cab1[v]);
      float ebb0 = __expf(2.0f * cbb0[v]);
      float ebb1 = __expf(2.0f * cbb1[v]);
      rsA0[v]  += eaa0;  rsA1[v]  += eaa1;
      rsAB0[v] += eab0;  rsAB1[v] += eab1;
      rsB0[v]  += ebb0;  rsB1[v]  += ebb1;
      colpart += eab0 + eab1;
      if (Rg0 == Cg) dab[Rg0] = eab0;  // numerator term exp(ab_ii/tau)
      if (Rg1 == Cg) dab[Rg1] = eab1;
    }
    colpart += __shfl_xor(colpart, 16, 32);  // combine row halves (same column)
    if (h == 0) atomicAdd(&sCol[Cg], colpart);
  }

  // finalize per-row sums: reduce across the 16 lanes of each half
#pragma unroll
  for (int v = 0; v < 8; ++v) {
    float a0 = rsA0[v],  a1 = rsA1[v];
    float p0 = rsAB0[v], p1 = rsAB1[v];
    float b0 = rsB0[v],  b1 = rsB1[v];
#pragma unroll
    for (int off = 1; off < 16; off <<= 1) {
      a0 += __shfl_xor(a0, off, 32);  a1 += __shfl_xor(a1, off, 32);
      p0 += __shfl_xor(p0, off, 32);  p1 += __shfl_xor(p1, off, 32);
      b0 += __shfl_xor(b0, off, 32);  b1 += __shfl_xor(b1, off, 32);
    }
    if (l15 == 0) {
      int Rg0 = r0 + v + 8 * h;
      int Rg1 = Rg0 + 16;
      atomicAdd(&rsaa[Rg0], a0);  atomicAdd(&rsaa[Rg1], a1);
      atomicAdd(&rsab[Rg0], p0);  atomicAdd(&rsab[Rg1], p1);
      atomicAdd(&rsbb[Rg0], b0);  atomicAdd(&rsbb[Rg1], b1);
    }
  }

  __syncthreads();
  for (int i = tid; i < NROWS; i += 128) atomicAdd(&csab[i], sCol[i]);
}

// ---------------- final loss reduction ----------------
__global__ __launch_bounds__(256) void final_loss(
    const float* __restrict__ rsaa, const float* __restrict__ rsab,
    const float* __restrict__ csab, const float* __restrict__ rsbb,
    const float* __restrict__ dab, float* __restrict__ out) {
  int t = threadIdx.x;
  const float e2 = __expf(2.0f);  // exp(||a||^2 / tau) == exp(2): diag of refl
  float acc = 0.0f;
  for (int i = t; i < NROWS; i += 256) {
    float num  = dab[i];
    float den1 = rsaa[i] + rsab[i] - e2;
    float den2 = rsbb[i] + csab[i] - e2;
    float l1 = __logf(den1) - __logf(num);
    float l2 = __logf(den2) - __logf(num);
    acc += 0.5f * (l1 + l2);
  }
  __shared__ float red[256];
  red[t] = acc;
  __syncthreads();
  for (int s = 128; s > 0; s >>= 1) {
    if (t < s) red[t] += red[t + s];
    __syncthreads();
  }
  if (t == 0) out[0] = red[0] / (float)NROWS;
}

// ---------------- launcher ----------------
extern "C" void kernel_launch(void* const* d_in, const int* in_sizes, int n_in,
                              void* d_out, int out_size, void* d_ws, size_t ws_size,
                              hipStream_t stream) {
  const float* z1 = (const float*)d_in[0];  // 8192x256
  const float* z2 = (const float*)d_in[1];  // 8192x256
  const float* W1 = (const float*)d_in[2];  // 256x256
  const float* b1 = (const float*)d_in[3];  // 256
  const float* W2 = (const float*)d_in[4];  // 256x256
  const float* b2 = (const float*)d_in[5];  // 256
  float* out = (float*)d_out;

  char* w = (char*)d_ws;
  size_t off = 0;
  auto take = [&](size_t bytes) { char* p = w + off; off += (bytes + 255) & ~size_t(255); return p; };

  __bf16* zb1 = (__bf16*)take((size_t)NROWS * DIM * 2);
  __bf16* zb2 = (__bf16*)take((size_t)NROWS * DIM * 2);
  __bf16* W1b = (__bf16*)take((size_t)DIM * DIM * 2);
  __bf16* W2b = (__bf16*)take((size_t)DIM * DIM * 2);
  __bf16* H1b = (__bf16*)take((size_t)NROWS * DIM * 2);
  __bf16* H2b = (__bf16*)take((size_t)NROWS * DIM * 2);
  float*  G1  = (float*)take((size_t)NROWS * DIM * 4);
  float*  G2  = (float*)take((size_t)NROWS * DIM * 4);
  __bf16* Ab  = (__bf16*)take((size_t)NROWS * DIM * 2);
  __bf16* Bb  = (__bf16*)take((size_t)NROWS * DIM * 2);
  float*  sums = (float*)take((size_t)5 * NROWS * 4);  // rsaa | rsab | csab | rsbb | dab
  float* rsaa = sums;
  float* rsab = sums + NROWS;
  float* csab = sums + 2 * NROWS;
  float* rsbb = sums + 3 * NROWS;
  float* dab  = sums + 4 * NROWS;

  const int nZ = NROWS * DIM, nW = DIM * DIM;

  // 1) convert inputs to bf16
  cvt_f32_bf16<<<(nZ + 255) / 256, 256, 0, stream>>>(z1, zb1, nZ);
  cvt_f32_bf16<<<(nZ + 255) / 256, 256, 0, stream>>>(z2, zb2, nZ);
  cvt_f32_bf16<<<(nW + 255) / 256, 256, 0, stream>>>(W1, W1b, nW);
  cvt_f32_bf16<<<(nW + 255) / 256, 256, 0, stream>>>(W2, W2b, nW);

  // 2) H = elu(z @ W1^T + b1)   (bf16 out)
  proj_gemm<true, true><<<dim3(NROWS / 32, DIM / 32, 2), 128, 0, stream>>>(
      zb1, zb2, W1b, b1, (void*)H1b, (void*)H2b);

  // 3) G = H @ W2^T + b2        (f32 out)
  proj_gemm<false, false><<<dim3(NROWS / 32, DIM / 32, 2), 128, 0, stream>>>(
      H1b, H2b, W2b, b2, (void*)G1, (void*)G2);

  // 4) row-normalize -> bf16 A, B
  normalize_rows<<<dim3(NROWS, 2), 256, 0, stream>>>(G1, G2, Ab, Bb);

  // 5) zero accumulators (must be done every launch: determinism)
  zero_f32<<<(5 * NROWS + 255) / 256, 256, 0, stream>>>(sums, 5 * NROWS);

  // 6) similarity matrices + exp + row/col sums (dominant: ~103 GFLOP)
  sim_kernel<<<NROWS / 32, 128, 0, stream>>>(Ab, Bb, rsaa, rsab, csab, rsbb, dab);

  // 7) final scalar loss
  final_loss<<<1, 256, 0, stream>>>(rsaa, rsab, csab, rsbb, dab, out);
}
